// TransformerLayer_7447473291398
// MI455X (gfx1250) — matrix operand order repaired and checked by reading, not compiled, not run on hardware
//
#include <hip/hip_runtime.h>
#include <hip/hip_bf16.h>

#define DI __device__ __forceinline__

typedef __bf16 bf16;
typedef __attribute__((ext_vector_type(16))) __bf16 v16bf;
typedef __attribute__((ext_vector_type(8)))  float  v8f;
typedef __attribute__((ext_vector_type(4)))  unsigned v4u;
typedef __attribute__((ext_vector_type(8)))  int      v8i;
typedef __attribute__((ext_vector_type(4)))  int      v4i;

static constexpr int Bz  = 2, T = 2048, C = 1024, H = 16, HS = 64, FF = 4096;
static constexpr int BT  = Bz * T;        // 4096 rows
static constexpr int QKVN = 3 * C;        // 3072
static constexpr float SCALE = 0.125f;    // HS^-0.5
static constexpr float NEGF  = -1e30f;

union AFragU { v16bf v; uint4 q[2]; };
union BFragU { v16bf v; uint4 q[2]; bf16 h[16]; };
union Pack8  { bf16 h[8]; uint4 q; };

DI v8f wmma_bf16(v16bf a, v16bf b, v8f c) {
  // D = A(16x32) * B(32x16) + C, f32 accum
  return __builtin_amdgcn_wmma_f32_16x16x32_bf16(false, a, false, b, (short)0, c, false, false);
}

// ---- A fragment: 16x32 (MxK) bf16 from row-major source, contiguous k = 16*half + s layout.
DI v16bf load_a_frag(const bf16* __restrict__ A, int lda, int m0, int k0, int lane) {
  int m = m0 + (lane & 15), half = lane >> 4;
  const bf16* p = A + (size_t)m * lda + k0 + half * 16;
  AFragU u;
  u.q[0] = *(const uint4*)(p);
  u.q[1] = *(const uint4*)(p + 8);
  return u.v;
}

// ---- B fragment 32x16 (KxN) with B[k][n] = S[n0+n][k0+k] (transposed read of row-major S)
DI v16bf load_bT_frag(const bf16* __restrict__ S, int ldS, int n0, int k0, int lane) {
  int n = n0 + (lane & 15), hl = lane >> 4;
  const bf16* p = S + (size_t)n * ldS + k0 + hl * 16;
  BFragU u;
  u.q[0] = *(const uint4*)(p);
  u.q[1] = *(const uint4*)(p + 8);
  return u.v;
}

// ===================== Tensor Data Mover: strided 2D tile -> LDS =====================
// D# per CDNA5 ISA ch.8: group0 = {flags(count=1), lds_addr, global_addr, type=2};
// group1 = {data_size=1(2B), tensor_dim0/1, tile_dim0/1, tensor_dim0_stride(48b)}.
// len/rows/stride in bf16 elements. Uniform (SGPR) operands; tracked on TENSORcnt.
DI void tdm_load_2d(unsigned ldsOff, const void* g, unsigned len, unsigned rows,
                    unsigned long long strideElems) {
  unsigned long long ga = (unsigned long long)(size_t)g;
  v4u g0;
  g0[0] = 1u;                                            // count=1, user descriptor
  g0[1] = ldsOff;                                        // lds_addr (bytes)
  g0[2] = (unsigned)(ga & 0xFFFFFFFFu);                  // global_addr[31:0]
  g0[3] = (unsigned)((ga >> 32) & 0x01FFFFFFu) | 0x80000000u; // addr[56:32] | type=2
  v8i g1;
  g1[0] = (int)(1u << 16);                               // wg_mask=0, data_size=1 (2B)
  g1[1] = (int)((len & 0xFFFFu) << 16);                  // tensor_dim0[15:0]
  g1[2] = (int)(((len >> 16) & 0xFFFFu) | ((rows & 0xFFFFu) << 16));  // dim0 hi | dim1 lo
  g1[3] = (int)(((rows >> 16) & 0xFFFFu) | ((len & 0xFFFFu) << 16));  // dim1 hi | tile_dim0
  g1[4] = (int)(rows & 0xFFFFu);                         // tile_dim1 (tile_dim2=0)
  g1[5] = (int)(unsigned)(strideElems & 0xFFFFFFFFull);  // dim0_stride[31:0]
  g1[6] = (int)(unsigned)((strideElems >> 32) & 0xFFFFull); // dim0_stride[47:32]
  g1[7] = 0;
  v4i z4 = {0, 0, 0, 0};
  v8i z8 = {0, 0, 0, 0, 0, 0, 0, 0};
  __builtin_amdgcn_tensor_load_to_lds(g0, g1, z4, z4, z8, 0);
}

// ===================== weight packing =====================
__global__ void pack_b_kernel(const float* __restrict__ W, bf16* __restrict__ out, int K, int N) {
  size_t t = (size_t)blockIdx.x * 256 + threadIdx.x;
  int i = (int)(t & 15); int lane = (int)((t >> 4) & 31); size_t rest = t >> 9;
  int ksteps = K >> 5;
  int ks = (int)(rest % (size_t)ksteps); int nt = (int)(rest / (size_t)ksteps);
  int k = ks * 32 + ((lane >> 4) << 4) + i;
  int n = nt * 16 + (lane & 15);
  out[t] = (bf16)W[(size_t)k * N + n];
}

__global__ void pack_b_qkv_kernel(const float* __restrict__ Wq, const float* __restrict__ Wk,
                                  const float* __restrict__ Wv, bf16* __restrict__ out) {
  size_t t = (size_t)blockIdx.x * 256 + threadIdx.x;
  int i = (int)(t & 15); int lane = (int)((t >> 4) & 31); size_t rest = t >> 9;
  const int ksteps = C >> 5; // 32
  int ks = (int)(rest % (size_t)ksteps); int nt = (int)(rest / (size_t)ksteps);
  int k = ks * 32 + ((lane >> 4) << 4) + i;
  int col = nt * 16 + (lane & 15);
  int proj = col >> 10, within = col & 1023;
  int hh = within >> 6, d = within & 63;
  const float* src = (proj == 0) ? Wq : ((proj == 1) ? Wk : Wv);
  out[t] = (bf16)src[((size_t)hh * C + k) * HS + d];
}

__global__ void pack_bias_qkv_kernel(const float* __restrict__ bq, const float* __restrict__ bk,
                                     const float* __restrict__ bv, float* __restrict__ out) {
  int t = blockIdx.x * 256 + threadIdx.x;
  if (t >= QKVN) return;
  int proj = t >> 10, rest = t & 1023;
  const float* src = (proj == 0) ? bq : ((proj == 1) ? bk : bv);
  out[t] = src[(rest >> 6) * HS + (rest & 63)];
}

// ===================== layernorm (fp32 in -> bf16 out) =====================
__global__ __launch_bounds__(256) void layernorm_kernel(const float* __restrict__ x,
                                                        const float* __restrict__ w,
                                                        bf16* __restrict__ out) {
  __shared__ float s1[256], s2[256];
  int row = blockIdx.x, tid = threadIdx.x;
  const float* xr = x + (size_t)row * C;
  float v[4]; float sum = 0.f, sq = 0.f;
#pragma unroll
  for (int i = 0; i < 4; ++i) { v[i] = xr[tid + i * 256]; sum += v[i]; sq += v[i] * v[i]; }
  s1[tid] = sum; s2[tid] = sq; __syncthreads();
  for (int off = 128; off > 0; off >>= 1) {
    if (tid < off) { s1[tid] += s1[tid + off]; s2[tid] += s2[tid + off]; }
    __syncthreads();
  }
  float mu = s1[0] * (1.0f / C);
  float var = s2[0] * (1.0f / C) - mu * mu;
  float rs = rsqrtf(var + 1e-5f);
  bf16* orow = out + (size_t)row * C;
#pragma unroll
  for (int i = 0; i < 4; ++i) { int c = tid + i * 256; orow[c] = (bf16)((v[i] - mu) * rs * w[c]); }
}

// ===================== WMMA GEMM with TDM double-buffered LDS pipeline =====================
// block: 256 thr = 8 waves (2 m-waves x 4 n-waves); block tile 64m x 256n; k-step 32.
// Wave0 issues tensor_load_to_lds for the A panel (64x32) and packed-B panel
// (16 rows of 512 elems, row stride (K/32)*512) into the next LDS buffer while
// all 8 waves run WMMA from the current buffer via ds_load_b128 fragment reads.
template<bool RELU, bool RESID, bool OUT_BF16>
__global__ __launch_bounds__(256) void gemm_kernel(const bf16* __restrict__ A,
                                                   const bf16* __restrict__ Bp,
                                                   const float* __restrict__ bias,
                                                   const float* __restrict__ resid,
                                                   bf16* __restrict__ outB,
                                                   float* __restrict__ outF,
                                                   int M, int N, int K) {
  __shared__ __align__(16) bf16 Ab[2][64 * 32];    // 8 KB
  __shared__ __align__(16) bf16 Bb[2][16 * 512];   // 32 KB
  int lane = threadIdx.x & 31, w = threadIdx.x >> 5;
  int mw = w & 1, nw = w >> 1;
  int m0 = blockIdx.y * 64;
  int ntBlock = blockIdx.x * 16;
  (void)M;
  int ksteps = K >> 5;

  const bf16* Abase  = A + (size_t)m0 * K;
  const bf16* Bpbase = Bp + (((size_t)ntBlock * (size_t)(K >> 5)) << 9);

  if (w == 0) {
    tdm_load_2d((unsigned)(size_t)&Ab[0][0], Abase, 32u, 64u, (unsigned long long)K);
    tdm_load_2d((unsigned)(size_t)&Bb[0][0], Bpbase, 512u, 16u, ((unsigned long long)(K >> 5)) << 9);
    __builtin_amdgcn_s_wait_tensorcnt(0);
  }
  __syncthreads();

  v8f zero = {};
  v8f acc[2][4];
#pragma unroll
  for (int mf = 0; mf < 2; ++mf)
#pragma unroll
    for (int nf = 0; nf < 4; ++nf) acc[mf][nf] = zero;

  int nl = lane & 15, hl = lane >> 4;
  for (int ks = 0; ks < ksteps; ++ks) {
    int cur = ks & 1;
    if (w == 0 && ks + 1 < ksteps) {
      tdm_load_2d((unsigned)(size_t)&Ab[cur ^ 1][0], Abase + (size_t)(ks + 1) * 32,
                  32u, 64u, (unsigned long long)K);
      tdm_load_2d((unsigned)(size_t)&Bb[cur ^ 1][0], Bpbase + ((size_t)(ks + 1) << 9),
                  512u, 16u, ((unsigned long long)(K >> 5)) << 9);
    }
    // A fragments from LDS (row-major 64x32 panel)
    v16bf a[2];
#pragma unroll
    for (int mf = 0; mf < 2; ++mf) {
      const bf16* p = &Ab[cur][(mw * 32 + mf * 16 + nl) * 32 + hl * 16];
      AFragU u;
      u.q[0] = *(const uint4*)(p);
      u.q[1] = *(const uint4*)(p + 8);
      a[mf] = u.v;
    }
#pragma unroll
    for (int nf = 0; nf < 4; ++nf) {
      const bf16* p = &Bb[cur][(nw * 4 + nf) * 512 + lane * 16];
      BFragU u;
      u.q[0] = *(const uint4*)(p);
      u.q[1] = *(const uint4*)(p + 8);
      acc[0][nf] = wmma_bf16(a[0], u.v, acc[0][nf]);
      acc[1][nf] = wmma_bf16(a[1], u.v, acc[1][nf]);
    }
    if (w == 0 && ks + 1 < ksteps) __builtin_amdgcn_s_wait_tensorcnt(0);
    __syncthreads();
  }

  // epilogue
#pragma unroll
  for (int mf = 0; mf < 2; ++mf)
#pragma unroll
    for (int nf = 0; nf < 4; ++nf) {
      int n = (ntBlock + nw * 4 + nf) * 16 + nl;
      float bi = bias ? bias[n] : 0.0f;
#pragma unroll
      for (int r = 0; r < 8; ++r) {
        int m = m0 + mw * 32 + mf * 16 + hl * 8 + r;
        float v = acc[mf][nf][r] + bi;
        if (RELU) v = fmaxf(v, 0.0f);
        if (RESID) v += resid[(size_t)m * N + n];
        if (OUT_BF16) outB[(size_t)m * N + n] = (bf16)v;
        else          outF[(size_t)m * N + n] = v;
      }
    }
}

// ===================== attention pass 1: per-KEY-column softmax stats =====================
__global__ __launch_bounds__(32) void attn_stats_kernel(const bf16* __restrict__ qkv,
                                                        float* __restrict__ mArr,
                                                        float* __restrict__ zArr) {
  int lane = threadIdx.x & 31;
  int jt = blockIdx.x & 127, bh = blockIdx.x >> 7;
  int b = bh >> 4, h = bh & 15;
  const bf16* qp = qkv + (size_t)b * T * QKVN + h * HS;
  const bf16* kp = qp + C;
  v16bf bk0 = load_bT_frag(kp, QKVN, jt * 16, 0, lane);   // B[k=d][n=j] = K[j][d]
  v16bf bk1 = load_bT_frag(kp, QKVN, jt * 16, 32, lane);
  int hl = lane >> 4, nl = lane & 15;
  int jg = jt * 16 + nl;
  float M = NEGF, Z = 0.0f;
  for (int it = jt; it < T / 16; ++it) {
    v16bf a0 = load_a_frag(qp, QKVN, it * 16, 0, lane);   // A[m=i][k=d]
    v16bf a1 = load_a_frag(qp, QKVN, it * 16, 32, lane);
    v8f c = {};
    c = wmma_bf16(a0, bk0, c);
    c = wmma_bf16(a1, bk1, c);
    float s[8]; bool val[8]; float mt = NEGF;
#pragma unroll
    for (int r = 0; r < 8; ++r) {
      int ig = it * 16 + hl * 8 + r;
      val[r] = (ig >= jg);
      s[r] = val[r] ? c[r] * SCALE : NEGF;
      mt = fmaxf(mt, s[r]);
    }
    float nM = fmaxf(M, mt);
    float zs = Z * __expf(M - nM);
#pragma unroll
    for (int r = 0; r < 8; ++r)
      if (val[r]) zs += __expf(s[r] - nM);
    M = nM; Z = zs;
  }
  float Mo = __shfl_xor(M, 16, 32);
  float Zo = __shfl_xor(Z, 16, 32);
  float Mc = fmaxf(M, Mo);
  float Zc = Z * __expf(M - Mc) + Zo * __expf(Mo - Mc);
  if (lane < 16) {
    mArr[(size_t)bh * T + jg] = Mc;
    zArr[(size_t)bh * T + jg] = Zc;
  }
}

// ===================== attention pass 2: att^T = V^T * P, P[j,i]=exp(s-m_j)/Z_j =====================
__global__ __launch_bounds__(32) void attn_av_kernel(const bf16* __restrict__ qkv,
                                                     const float* __restrict__ mArr,
                                                     const float* __restrict__ zArr,
                                                     bf16* __restrict__ att) {
  __shared__ __align__(16) bf16 ptile[32 * 16];   // P[j(32)][i(16)]
  __shared__ __align__(16) bf16 vtile[32 * 64];   // V[j(32)][d(64)]
  int lane = threadIdx.x & 31;
  int it = blockIdx.x & 127, bh = blockIdx.x >> 7;
  int b = bh >> 4, h = bh & 15;
  const bf16* qp = qkv + (size_t)b * T * QKVN + h * HS;
  const bf16* kp = qp + C;
  const bf16* vp = qp + 2 * C;
  int hl = lane >> 4, nl = lane & 15;
  v16bf bq0 = load_bT_frag(qp, QKVN, it * 16, 0, lane);   // B[k=d][n=i] = Q[i][d]
  v16bf bq1 = load_bT_frag(qp, QKVN, it * 16, 32, lane);
  int ig = it * 16 + nl;
  const float* mrow = mArr + (size_t)bh * T;
  const float* zrow = zArr + (size_t)bh * T;
  v8f zero = {};
  v8f acc[4];
#pragma unroll
  for (int dt = 0; dt < 4; ++dt) acc[dt] = zero;

  for (int jt2 = 0; jt2 <= it; jt2 += 2) {
    {
      const bf16* src = vp + (size_t)(jt2 * 16 + lane) * QKVN;
#pragma unroll
      for (int cch = 0; cch < 8; ++cch)
        *(uint4*)(vtile + lane * 64 + cch * 8) = *(const uint4*)(src + cch * 8);
    }
#pragma unroll
    for (int s = 0; s < 2; ++s) {
      int jt = jt2 + s;
      v16bf ak0 = load_a_frag(kp, QKVN, jt * 16, 0, lane);  // A[m=j][k=d]
      v16bf ak1 = load_a_frag(kp, QKVN, jt * 16, 32, lane);
      v8f c = {};
      c = wmma_bf16(ak0, bq0, c);
      c = wmma_bf16(ak1, bq1, c);
#pragma unroll
      for (int r = 0; r < 8; ++r) {
        int jg = jt * 16 + hl * 8 + r;
        float p = 0.0f;
        if (ig >= jg) {
          float mj = mrow[jg], zj = zrow[jg];
          p = __expf(c[r] * SCALE - mj) / zj;
        }
        ptile[(s * 16 + hl * 8 + r) * 16 + nl] = (bf16)p;
      }
    }
    __syncthreads();
    BFragU bpf;   // B[k=j][n=i] from LDS
#pragma unroll
    for (int e = 0; e < 16; ++e) bpf.h[e] = ptile[(hl * 16 + e) * 16 + nl];
#pragma unroll
    for (int dt = 0; dt < 4; ++dt) {
      BFragU av;  // A[m=d][k=j] = V[j][d]
#pragma unroll
      for (int e = 0; e < 16; ++e) {
        int jr = hl * 16 + e;
        av.h[e] = vtile[jr * 64 + dt * 16 + nl];
      }
      acc[dt] = wmma_bf16(av.v, bpf.v, acc[dt]);
    }
    __syncthreads();
  }
  bf16* orow = att + ((size_t)(b * T) + (size_t)(it * 16 + nl)) * C + h * HS;
#pragma unroll
  for (int dt = 0; dt < 4; ++dt) {
    Pack8 pk;
#pragma unroll
    for (int r = 0; r < 8; ++r) pk.h[r] = (bf16)acc[dt][r];
    *(uint4*)(orow + dt * 16 + hl * 8) = pk.q;
  }
}

// ===================== launch =====================
extern "C" void kernel_launch(void* const* d_in, const int* in_sizes, int n_in,
                              void* d_out, int out_size, void* d_ws, size_t ws_size,
                              hipStream_t stream) {
  (void)in_sizes; (void)n_in; (void)out_size; (void)ws_size;
  const float* x    = (const float*)d_in[0];
  const float* ln1w = (const float*)d_in[1];
  const float* Wq   = (const float*)d_in[2];
  const float* bq   = (const float*)d_in[3];
  const float* Wk   = (const float*)d_in[4];
  const float* bk   = (const float*)d_in[5];
  const float* Wv   = (const float*)d_in[6];
  const float* bv   = (const float*)d_in[7];
  const float* Wo   = (const float*)d_in[8];
  const float* bo   = (const float*)d_in[9];
  const float* ln2w = (const float*)d_in[10];
  const float* W1   = (const float*)d_in[11];
  const float* b1   = (const float*)d_in[12];
  const float* W2   = (const float*)d_in[13];
  const float* b2   = (const float*)d_in[14];

  size_t off = 0;
  auto carve = [&](size_t bytes) -> void* {
    void* p = (char*)d_ws + off;
    off += (bytes + 255) & ~(size_t)255;
    return p;
  };
  bf16*  BpQKV  = (bf16*)carve((size_t)C * QKVN * 2);
  bf16*  BpO    = (bf16*)carve((size_t)C * C * 2);
  bf16*  Bp1    = (bf16*)carve((size_t)C * FF * 2);
  bf16*  Bp2    = (bf16*)carve((size_t)FF * C * 2);
  float* biasQKV= (float*)carve((size_t)QKVN * 4);
  bf16*  hA     = (bf16*)carve((size_t)BT * C * 2);
  bf16*  qkv    = (bf16*)carve((size_t)BT * QKVN * 2);
  float* mArr   = (float*)carve((size_t)Bz * H * T * 4);
  float* zArr   = (float*)carve((size_t)Bz * H * T * 4);
  bf16*  attB   = (bf16*)carve((size_t)BT * C * 2);
  float* x1     = (float*)carve((size_t)BT * C * 4);
  bf16*  h2     = (bf16*)carve((size_t)BT * C * 2);
  bf16*  ff1    = (bf16*)carve((size_t)BT * FF * 2);

  pack_b_qkv_kernel<<<(C * QKVN) / 256, 256, 0, stream>>>(Wq, Wk, Wv, BpQKV);
  pack_b_kernel<<<(C * C) / 256, 256, 0, stream>>>(Wo, BpO, C, C);
  pack_b_kernel<<<(C * FF) / 256, 256, 0, stream>>>(W1, Bp1, C, FF);
  pack_b_kernel<<<(FF * C) / 256, 256, 0, stream>>>(W2, Bp2, FF, C);
  pack_bias_qkv_kernel<<<(QKVN + 255) / 256, 256, 0, stream>>>(bq, bk, bv, biasQKV);

  layernorm_kernel<<<BT, 256, 0, stream>>>(x, ln1w, hA);

  gemm_kernel<false, false, true><<<dim3(QKVN / 256, BT / 64), 256, 0, stream>>>(
      hA, BpQKV, biasQKV, nullptr, qkv, nullptr, BT, QKVN, C);

  attn_stats_kernel<<<Bz * H * (T / 16), 32, 0, stream>>>(qkv, mArr, zArr);
  attn_av_kernel<<<Bz * H * (T / 16), 32, 0, stream>>>(qkv, mArr, zArr, attB);

  gemm_kernel<false, true, false><<<dim3(C / 256, BT / 64), 256, 0, stream>>>(
      attB, BpO, bo, x, nullptr, x1, BT, C, C);

  layernorm_kernel<<<BT, 256, 0, stream>>>(x1, ln2w, h2);

  gemm_kernel<true, false, true><<<dim3(FF / 256, BT / 64), 256, 0, stream>>>(
      h2, Bp1, b1, nullptr, ff1, nullptr, BT, FF, C);

  gemm_kernel<false, true, false><<<dim3(C / 256, BT / 64), 256, 0, stream>>>(
      ff1, Bp2, b2, x1, nullptr, (float*)d_out, BT, C, FF);
}
